// RelGraphConv_49821620633797
// MI455X (gfx1250) — matrix operand order repaired and checked by reading, not compiled, not run on hardware
//
#include <hip/hip_runtime.h>
#include <hip/hip_bf16.h>

// RGCN (basis decomposition) for MI455X / gfx1250, wave32.
//
//   coeff[e,b] = norm[e] * w_comp[etype[e], b]
//   t_b[d]     = sum_{e: dst[e]==d} coeff[e,b] * feat[src[e]]     (CSR, no float atomics)
//   X          = [t_0 .. t_7 | feat]                  (N x 2304, bf16, padded to 50016 rows)
//   WcatT      = [weight_0; ..; weight_7; loop_weight]^T (256 x 2304, bf16, L2-resident)
//   out        = relu(X @ Wcat + bias)                (bf16 WMMA, fp32 accumulate)
//
// Workspace: ~261 MB (X 230.5MB, coeff 25.6MB, CSR ~3.6MB, WcatT 1.2MB).

typedef __attribute__((ext_vector_type(16))) __bf16 v16bf;
typedef __attribute__((ext_vector_type(8)))  __bf16 v8bf;
typedef __attribute__((ext_vector_type(8)))  float  v8f;

#define N_NODES 50000
#define N_PAD   50016                     // next multiple of 32 (pad rows zeroed)
#define N_EDGES 800000
#define IN_F    256
#define OUT_F   256
#define N_REL   16
#define N_BASES 8
#define KDIM    (N_BASES * IN_F + IN_F)   // 2304

// ---------------------------------------------------------------- CSR build
__global__ void k_count_coeff(const int* __restrict__ dst, const int* __restrict__ etyp,
                              const float* __restrict__ norm, const float* __restrict__ w_comp,
                              int* __restrict__ cnt, float* __restrict__ coeff) {
    int e = blockIdx.x * blockDim.x + threadIdx.x;
    if (e >= N_EDGES) return;
    atomicAdd(&cnt[dst[e]], 1);
    float c = norm[e];
    int   r = etyp[e];
#pragma unroll
    for (int b = 0; b < N_BASES; ++b)
        coeff[(size_t)e * N_BASES + b] = c * w_comp[r * N_BASES + b];
}

__global__ void k_scan(const int* __restrict__ cnt, int* __restrict__ row_ptr) {
    __shared__ int sums[1024];
    const int T = 1024;
    const int chunk = (N_NODES + T - 1) / T;        // 49
    int t   = threadIdx.x;
    int beg = t * chunk;
    int end = beg + chunk; if (end > N_NODES) end = N_NODES;
    int s = 0;
    for (int i = beg; i < end && i < N_NODES; ++i) s += cnt[i];
    sums[t] = s;
    __syncthreads();
    if (t == 0) {
        int run = 0;
        for (int i = 0; i < T; ++i) { int v = sums[i]; sums[i] = run; run += v; }
    }
    __syncthreads();
    int run = sums[t];
    for (int i = beg; i < end && i < N_NODES; ++i) { row_ptr[i] = run; run += cnt[i]; }
    if (end == N_NODES) row_ptr[N_NODES] = run;     // all such threads write identical total
}

__global__ void k_fill(const int* __restrict__ dst, const int* __restrict__ row_ptr,
                       int* __restrict__ cur, int* __restrict__ eids) {
    int e = blockIdx.x * blockDim.x + threadIdx.x;
    if (e >= N_EDGES) return;
    int d   = dst[e];
    int pos = row_ptr[d] + atomicAdd(&cur[d], 1);
    eids[pos] = e;
}

// ------------------------------------------------------- per-dst aggregation
__global__ __launch_bounds__(256) void k_aggregate(
        const int* __restrict__ row_ptr, const int* __restrict__ eids,
        const int* __restrict__ src, const float* __restrict__ coeff,
        const float* __restrict__ feat, __bf16* __restrict__ X) {
    int n = blockIdx.x;
    int i = threadIdx.x;
    int beg = row_ptr[n], end = row_ptr[n + 1];
    float acc[N_BASES] = {0.f, 0.f, 0.f, 0.f, 0.f, 0.f, 0.f, 0.f};
    for (int j = beg; j < end; ++j) {
        int e = eids[j];
        int s = src[e];
        float f = feat[(size_t)s * IN_F + i];                 // coalesced 1KB row
        const float4* cp = (const float4*)(coeff + (size_t)e * N_BASES);
        float4 c0 = cp[0], c1 = cp[1];                        // broadcast (cache hit)
        acc[0] += c0.x * f; acc[1] += c0.y * f; acc[2] += c0.z * f; acc[3] += c0.w * f;
        acc[4] += c1.x * f; acc[5] += c1.y * f; acc[6] += c1.z * f; acc[7] += c1.w * f;
    }
    __bf16* xr = X + (size_t)n * KDIM;
#pragma unroll
    for (int b = 0; b < N_BASES; ++b) xr[b * IN_F + i] = (__bf16)acc[b];
    xr[N_BASES * IN_F + i] = (__bf16)feat[(size_t)n * IN_F + i];
}

// ------------------------------------------------- pack transposed bf16 Wcat
__global__ void k_pack_w(const float* __restrict__ weight, const float* __restrict__ loop_w,
                         __bf16* __restrict__ WT) {
    int idx = blockIdx.x * blockDim.x + threadIdx.x;
    if (idx >= OUT_F * KDIM) return;
    int nn = idx / KDIM, k = idx % KDIM;
    float v = (k < N_BASES * IN_F)
            ? weight[(size_t)k * OUT_F + nn]
            : loop_w[(size_t)(k - N_BASES * IN_F) * OUT_F + nn];
    WT[idx] = (__bf16)v;
}

// ------------------------------------------------------------- WMMA GEMM
// out = relu(X @ Wcat + bias).  Block = 256 threads = 8 waves.
// Wave owns a 32-row x 64-col strip: two A tiles x four B tiles = 8 WMMAs
// per 32-K stage, double-buffered over K (KDIM % 64 == 0 -> exact 2-stage
// pipe, no copies).  X is padded to 50016 rows (zeros), so every strip runs
// the full unconditional pipeline: no EXEC manipulation near any WMMA.
// blockIdx.x = column group (fastest varying) so the 4 blocks sharing the
// same X rows launch adjacently -> X rows hit the 192MB L2 instead of being
// re-streamed from HBM per column group.
__global__ __launch_bounds__(256) void k_gemm(
        const __bf16* __restrict__ X, const __bf16* __restrict__ WT,
        const float* __restrict__ bias, float* __restrict__ out) {
    const int lane  = threadIdx.x & 31;
    const int wave  = threadIdx.x >> 5;
    const int strip = blockIdx.y * 8 + wave;         // 32-row strip index
    const int m0    = strip * 32;
    if (m0 >= N_PAD) return;                         // wave-uniform guard
    const int n0    = blockIdx.x * 64;
    const int lmod  = lane & 15;
    const int lhi   = lane >> 4;

    // A operand (16x32 bf16): lane holds row m+lmod, K segs [s,s+8) & [s+16,s+24), s=lhi*8.
    const __bf16* arow0 = X + (size_t)(m0 + lmod) * KDIM + lhi * 8;
    const __bf16* arow1 = arow0 + (size_t)16 * KDIM;
    // B operand (32x16 bf16): lane holds column n0+t*16+lmod, K segs [s,s+8) & [s+16,s+24), s=lhi*8.
    const __bf16* bcol  = WT + (size_t)(n0 + lmod) * KDIM + lhi * 8;

    v8f c0[4] = {};                                  // rows m0..m0+15
    v8f c1[4] = {};                                  // rows m0+16..m0+31

    union AB { v16bf v; v8bf h[2]; };
    AB a0[2], a1[2];                                 // [stage]
    v16bf b[2][4];                                   // [stage][tile]

    // ---- prologue: stage 0 @ k = 0
    a0[0].h[0] = *(const v8bf*)(arow0);      a0[0].h[1] = *(const v8bf*)(arow0 + 16);
    a1[0].h[0] = *(const v8bf*)(arow1);      a1[0].h[1] = *(const v8bf*)(arow1 + 16);
#pragma unroll
    for (int t = 0; t < 4; ++t) b[0][t] = __builtin_shufflevector(*(const v8bf*)(bcol + (size_t)t * 16 * KDIM), *(const v8bf*)(bcol + (size_t)t * 16 * KDIM + 16), 0,1,2,3,4,5,6,7,8,9,10,11,12,13,14,15);

    for (int k = 0; k < KDIM; k += 64) {
        // ---- issue stage-1 loads @ k+32 (always valid: KDIM % 64 == 0)
        __builtin_prefetch(arow0 + k + 256, 0, 1);   // global_prefetch_b8 (A stream)
        a0[1].h[0] = *(const v8bf*)(arow0 + k + 32);
        a0[1].h[1] = *(const v8bf*)(arow0 + k + 48);
        a1[1].h[0] = *(const v8bf*)(arow1 + k + 32);
        a1[1].h[1] = *(const v8bf*)(arow1 + k + 48);
#pragma unroll
        for (int t = 0; t < 4; ++t)
            b[1][t] = __builtin_shufflevector(*(const v8bf*)(bcol + (size_t)t * 16 * KDIM + k + 32), *(const v8bf*)(bcol + (size_t)t * 16 * KDIM + k + 48), 0,1,2,3,4,5,6,7,8,9,10,11,12,13,14,15);

        // ---- consume stage 0
#pragma unroll
        for (int t = 0; t < 4; ++t)
            c0[t] = __builtin_amdgcn_wmma_f32_16x16x32_bf16(
                        false, a0[0].v, false, b[0][t], (short)0, c0[t], false, false);
#pragma unroll
        for (int t = 0; t < 4; ++t)
            c1[t] = __builtin_amdgcn_wmma_f32_16x16x32_bf16(
                        false, a1[0].v, false, b[0][t], (short)0, c1[t], false, false);

        // ---- issue stage-0 loads @ k+64
        if (k + 64 < KDIM) {
            __builtin_prefetch(arow1 + k + 256, 0, 1);
            a0[0].h[0] = *(const v8bf*)(arow0 + k + 64);
            a0[0].h[1] = *(const v8bf*)(arow0 + k + 80);
            a1[0].h[0] = *(const v8bf*)(arow1 + k + 64);
            a1[0].h[1] = *(const v8bf*)(arow1 + k + 80);
#pragma unroll
            for (int t = 0; t < 4; ++t)
                b[0][t] = __builtin_shufflevector(*(const v8bf*)(bcol + (size_t)t * 16 * KDIM + k + 64), *(const v8bf*)(bcol + (size_t)t * 16 * KDIM + k + 80), 0,1,2,3,4,5,6,7,8,9,10,11,12,13,14,15);
        }

        // ---- consume stage 1
#pragma unroll
        for (int t = 0; t < 4; ++t)
            c0[t] = __builtin_amdgcn_wmma_f32_16x16x32_bf16(
                        false, a0[1].v, false, b[1][t], (short)0, c0[t], false, false);
#pragma unroll
        for (int t = 0; t < 4; ++t)
            c1[t] = __builtin_amdgcn_wmma_f32_16x16x32_bf16(
                        false, a1[1].v, false, b[1][t], (short)0, c1[t], false, false);
    }

    // D layout (32-bit 16x16): lanes 0-15 -> N=lane, M=r; lanes 16-31 -> M=8+r.
    // Store guards are wave-uniform and only gate scalar stores (WMMA done).
#pragma unroll
    for (int t = 0; t < 4; ++t) {
        int col = n0 + t * 16 + lmod;
        float bv = bias[col];
#pragma unroll
        for (int r = 0; r < 8; ++r) {
            int row = m0 + r + 8 * lhi;
            float v = c0[t][r] + bv;
            out[(size_t)row * OUT_F + col] = v > 0.f ? v : 0.f;
        }
    }
    if (m0 + 16 < N_NODES) {
#pragma unroll
        for (int t = 0; t < 4; ++t) {
            int col = n0 + t * 16 + lmod;
            float bv = bias[col];
#pragma unroll
            for (int r = 0; r < 8; ++r) {
                int row = m0 + 16 + r + 8 * lhi;
                float v = c1[t][r] + bv;
                out[(size_t)row * OUT_F + col] = v > 0.f ? v : 0.f;
            }
        }
    }
}

// ---------------------------------------------------------------- launcher
extern "C" void kernel_launch(void* const* d_in, const int* in_sizes, int n_in,
                              void* d_out, int out_size, void* d_ws, size_t ws_size,
                              hipStream_t stream) {
    const float* feat   = (const float*)d_in[0];
    const float* norm   = (const float*)d_in[1];
    const float* weight = (const float*)d_in[2];   // (B, IN, OUT) row-major
    const float* w_comp = (const float*)d_in[3];   // (R, B)
    const float* h_bias = (const float*)d_in[4];   // (OUT,)
    const float* loop_w = (const float*)d_in[5];   // (IN, OUT)
    const int*   src    = (const int*)d_in[6];
    const int*   dst    = (const int*)d_in[7];
    const int*   etyp   = (const int*)d_in[8];
    float* out = (float*)d_out;

    char* ws = (char*)d_ws;
    size_t off = 0;
    auto carve = [&](size_t bytes) -> char* {
        char* p = ws + off;
        off = (off + bytes + 255) & ~(size_t)255;
        return p;
    };
    __bf16* X     = (__bf16*)carve((size_t)N_PAD * KDIM * sizeof(__bf16));     // 230.5 MB
    __bf16* WT    = (__bf16*)carve((size_t)OUT_F * KDIM * sizeof(__bf16));     // 1.2 MB
    float*  coeff = (float*) carve((size_t)N_EDGES * N_BASES * sizeof(float)); // 25.6 MB
    int* row_ptr  = (int*)   carve((size_t)(N_NODES + 1) * sizeof(int));
    int* cnt      = (int*)   carve((size_t)N_NODES * sizeof(int));
    int* eids     = (int*)   carve((size_t)N_EDGES * sizeof(int));

    hipMemsetAsync(cnt, 0, (size_t)N_NODES * sizeof(int), stream);
    // zero the 16 pad rows of X so the tail strip multiplies finite zeros
    hipMemsetAsync(X + (size_t)N_NODES * KDIM, 0,
                   (size_t)(N_PAD - N_NODES) * KDIM * sizeof(__bf16), stream);
    k_count_coeff<<<(N_EDGES + 255) / 256, 256, 0, stream>>>(dst, etyp, norm, w_comp, cnt, coeff);
    k_scan<<<1, 1024, 0, stream>>>(cnt, row_ptr);
    hipMemsetAsync(cnt, 0, (size_t)N_NODES * sizeof(int), stream);   // reuse as fill cursor
    k_fill<<<(N_EDGES + 255) / 256, 256, 0, stream>>>(dst, row_ptr, cnt, eids);
    k_aggregate<<<N_NODES, 256, 0, stream>>>(row_ptr, eids, src, coeff, feat, X);
    k_pack_w<<<(OUT_F * KDIM + 255) / 256, 256, 0, stream>>>(weight, loop_w, WT);

    // 1563 strips of 32 rows; 8 strips per block.
    // x = column group so blocks sharing X rows are launch-adjacent (L2 reuse).
    dim3 grid(OUT_F / 64, (N_PAD / 32 + 7) / 8);     // (4, 196)
    k_gemm<<<grid, 256, 0, stream>>>(X, WT, h_bias, out);
}
